// GCNEncoder_56607668961286
// MI455X (gfx1250) — matrix operand order repaired and checked by reading, not compiled, not run on hardware
//
#include <hip/hip_runtime.h>

typedef __attribute__((ext_vector_type(16))) __bf16 v16bf;
typedef __attribute__((ext_vector_type(8)))  float  v8f;

static __device__ __forceinline__ unsigned short f2bf(float f) {
    // round-to-nearest-even float -> bf16
    unsigned int u = __builtin_bit_cast(unsigned int, f);
    u += 0x7FFFu + ((u >> 16) & 1u);
    return (unsigned short)(u >> 16);
}

// ---------------- degree / norm ----------------
__global__ void k_fill1(float* __restrict__ deg, int n) {
    int i = blockIdx.x * blockDim.x + threadIdx.x;
    if (i < n) deg[i] = 1.0f;               // self-loop contribution
}

__global__ void k_count_deg(const int* __restrict__ col, float* __restrict__ deg, int E) {
    int i = blockIdx.x * blockDim.x + threadIdx.x;
    if (i < E) atomicAdd(&deg[col[i]], 1.0f);
}

__global__ void k_rsqrt(const float* __restrict__ deg, float* __restrict__ dinv, int n) {
    int i = blockIdx.x * blockDim.x + threadIdx.x;
    if (i < n) dinv[i] = rsqrtf(deg[i]);    // deg >= 1 always (self-loops)
}

// ---------------- bf16 conversion ----------------
__global__ void k_cvt_bf16(const float* __restrict__ x, unsigned short* __restrict__ y, int n) {
    int i = blockIdx.x * blockDim.x + threadIdx.x;
    if (i < n) y[i] = f2bf(x[i]);
}

// W is K x Ncol row-major fp32; produce Wt = Ncol x K row-major bf16
__global__ void k_cvt_transpose(const float* __restrict__ W, unsigned short* __restrict__ Wt,
                                int K, int Ncol) {
    int o = blockIdx.x * blockDim.x + threadIdx.x;
    if (o >= K * Ncol) return;
    int n = o / K;
    int k = o - n * K;
    Wt[o] = f2bf(W[(size_t)k * Ncol + n]);
}

// ---------------- WMMA GEMM: C[MxN] = A[MxK] * Bt[NxK]^T (bf16 in, f32 out) ----------------
// block = (32,4): 4 waves, each wave owns one 16-wide n-tile.
// REQUIRES: M % 16 == 0, Ncol % 64 == 0, K == 128 (both layers: K = 128).
__global__ void k_wmma_gemm_bf16(const unsigned short* __restrict__ A,
                                 const unsigned short* __restrict__ Bt,
                                 float* __restrict__ C,
                                 int Ncol, int K) {
    const int lane = threadIdx.x;          // 0..31
    const int kh   = lane >> 4;            // 0/1: which K-half this lane holds
    const int l16  = lane & 15;
    const int tn   = blockIdx.y * 4 + threadIdx.y;   // n-tile index
    const int n0   = tn * 16;
    const int m    = blockIdx.x * 16 + l16;

    const unsigned short* arow = A  + (size_t)m * K;
    const unsigned short* brow = Bt + (size_t)(n0 + l16) * K;

    union U { v16bf bf; uint4 q[2]; };
    v8f acc = {};

#pragma unroll
    for (int kb = 0; kb < 128; kb += 32) {
        U a, b;
        // A lane layout: elems 0..7 -> K = kb + kh*8 .. +7 ; elems 8..15 -> K = kb+16+kh*8 ..
        a.q[0] = *(const uint4*)(arow + kb + kh * 8);
        a.q[1] = *(const uint4*)(arow + kb + 16 + kh * 8);
        // B lane layout: elems 0..7 -> K = kb + kh*8 .. +7 ; elems 8..15 -> K = kb+16+kh*8 ..
        b.q[0] = *(const uint4*)(brow + kb + kh * 8);
        b.q[1] = *(const uint4*)(brow + kb + 16 + kh * 8);
        acc = __builtin_amdgcn_wmma_f32_16x16x32_bf16(false, a.bf, false, b.bf,
                                                      (short)0, acc, false, false);
    }

    // D layout: lane -> column n0 + l16; VGPR v -> row mbase + v, mbase = tile*16 + 8*kh.
    const int mbase = blockIdx.x * 16 + kh * 8;
    float* cp = C + (size_t)mbase * Ncol + n0 + l16;
#pragma unroll
    for (int v = 0; v < 8; ++v) {
        cp[(size_t)v * Ncol] = acc[v];     // unguarded: M % 16 == 0 by construction
    }
}

// ---------------- aggregation ----------------
// out[i,f] = h[i,f] * dinv[i]^2 (+ bias[f])   -- self-loop seed, fully initializes out
__global__ void k_self_init(const float* __restrict__ h, const float* __restrict__ dinv,
                            const float* __restrict__ bias, float* __restrict__ outp,
                            int total, int F) {
    int i = blockIdx.x * blockDim.x + threadIdx.x;
    if (i >= total) return;
    int node = i / F;
    int f = i - node * F;
    float di = dinv[node];
    float v = h[i] * di * di;
    if (bias) v += bias[f];
    outp[i] = v;
}

// per-edge scatter: agg[col] += norm * h[row], float4 per thread
__global__ void k_edge_scatter(const int* __restrict__ row, const int* __restrict__ col,
                               const float* __restrict__ dinv,
                               const float* __restrict__ h, float* __restrict__ agg,
                               int E, int F) {
    int tid = blockIdx.x * blockDim.x + threadIdx.x;
    int chunks = F >> 2;
    if (tid >= E * chunks) return;
    int e = tid / chunks;
    int c = tid - e * chunks;
    int r = row[e], t = col[e];
    float nrm = dinv[r] * dinv[t];
    const float4 v = *(const float4*)(h + (size_t)r * F + c * 4);
    float* dst = agg + (size_t)t * F + c * 4;
    atomicAdd(dst + 0, v.x * nrm);
    atomicAdd(dst + 1, v.y * nrm);
    atomicAdd(dst + 2, v.z * nrm);
    atomicAdd(dst + 3, v.w * nrm);
}

// y_bf16[i,f] = bf16(relu(agg[i,f] + bias[f]))
__global__ void k_bias_relu_cvt(const float* __restrict__ agg, const float* __restrict__ bias,
                                unsigned short* __restrict__ ybf, int total, int F) {
    int i = blockIdx.x * blockDim.x + threadIdx.x;
    if (i >= total) return;
    int f = i % F;
    float v = agg[i] + bias[f];
    ybf[i] = f2bf(v > 0.0f ? v : 0.0f);
}

static inline int cdiv(int a, int b) { return (a + b - 1) / b; }

extern "C" void kernel_launch(void* const* d_in, const int* in_sizes, int n_in,
                              void* d_out, int out_size, void* d_ws, size_t ws_size,
                              hipStream_t stream) {
    (void)n_in; (void)out_size; (void)ws_size;

    const float* x  = (const float*)d_in[0];
    const int*   ei = (const int*)  d_in[1];
    const float* W1 = (const float*)d_in[2];
    const float* b1 = (const float*)d_in[3];
    const float* W2 = (const float*)d_in[4];
    const float* b2 = (const float*)d_in[5];
    float* out = (float*)d_out;

    const int HID = in_sizes[3];            // 128
    const int OUT = in_sizes[5];            // 64
    const int IN  = in_sizes[2] / HID;      // 128
    const int N   = in_sizes[0] / IN;       // 50000 (divisible by 16)
    const int E   = in_sizes[1] / 2;        // 800000
    const int* row = ei;
    const int* col = ei + E;

    // bump allocator over d_ws (256B aligned chunks)
    char* p = (char*)d_ws;
    auto alloc = [&](size_t bytes) -> void* {
        void* r = (void*)p;
        p += (bytes + 255) & ~(size_t)255;
        return r;
    };
    float*          deg  = (float*)         alloc((size_t)N * 4);
    float*          dinv = (float*)         alloc((size_t)N * 4);
    unsigned short* xbf  = (unsigned short*)alloc((size_t)N * IN * 2);
    unsigned short* w1t  = (unsigned short*)alloc((size_t)HID * IN * 2);
    unsigned short* w2t  = (unsigned short*)alloc((size_t)OUT * HID * 2);
    float*          h1   = (float*)         alloc((size_t)N * HID * 4);
    float*          agg1 = (float*)         alloc((size_t)N * HID * 4);
    unsigned short* h1bf = xbf;             // reuse: xbf dead after GEMM1
    float*          h2   = h1;              // reuse: h1 dead after edge_scatter(1)

    const int T = 256;

    // --- normalization ---
    k_fill1    <<<cdiv(N, T), T, 0, stream>>>(deg, N);
    k_count_deg<<<cdiv(E, T), T, 0, stream>>>(col, deg, E);
    k_rsqrt    <<<cdiv(N, T), T, 0, stream>>>(deg, dinv, N);

    // --- bf16 operand prep ---
    k_cvt_bf16     <<<cdiv(N * IN, T), T, 0, stream>>>(x, xbf, N * IN);
    k_cvt_transpose<<<cdiv(IN * HID, T), T, 0, stream>>>(W1, w1t, IN, HID);
    k_cvt_transpose<<<cdiv(HID * OUT, T), T, 0, stream>>>(W2, w2t, HID, OUT);

    // --- layer 1: GEMM (WMMA bf16) + aggregate + bias/relu ---
    {
        dim3 grid(N / 16, HID / 64), blk(32, 4);
        k_wmma_gemm_bf16<<<grid, blk, 0, stream>>>(xbf, w1t, h1, HID, IN);
    }
    k_self_init   <<<cdiv(N * HID, T), T, 0, stream>>>(h1, dinv, nullptr, agg1, N * HID, HID);
    k_edge_scatter<<<cdiv(E * (HID / 4), T), T, 0, stream>>>(row, col, dinv, h1, agg1, E, HID);
    k_bias_relu_cvt<<<cdiv(N * HID, T), T, 0, stream>>>(agg1, b1, h1bf, N * HID, HID);

    // --- layer 2: GEMM (WMMA bf16) + aggregate + bias (into d_out) ---
    {
        dim3 grid(N / 16, OUT / 64), blk(32, 4);
        k_wmma_gemm_bf16<<<grid, blk, 0, stream>>>(h1bf, w2t, h2, OUT, HID);
    }
    k_self_init   <<<cdiv(N * OUT, T), T, 0, stream>>>(h2, dinv, b2, out, N * OUT, OUT);
    k_edge_scatter<<<cdiv(E * (OUT / 4), T), T, 0, stream>>>(row, col, dinv, h2, out, E, OUT);
}
